// EPMoE_20444044329134
// MI455X (gfx1250) — compile-verified, matrix operand order checked by reading, not run on hardware
//
#include <hip/hip_runtime.h>

#define N_EXPERTS 8
#define HIDDEN 1024
#define INTER 2048
#define T_TOKENS 1024

typedef __bf16 bf16_t;
typedef __attribute__((ext_vector_type(16))) __bf16 v16bf;
typedef __attribute__((ext_vector_type(8))) float v8f;
typedef __attribute__((ext_vector_type(4))) unsigned int v4u;
typedef __attribute__((ext_vector_type(8))) int v8i;
typedef __attribute__((ext_vector_type(4))) int v4i;

// ---------------------------------------------------------------------------
// Kernel 1: gating — softmax + grouped top-k -> combine[T][8]
// ---------------------------------------------------------------------------
__global__ __launch_bounds__(256) void gate_kernel(
    const float* __restrict__ x,      // [T, H]
    const float* __restrict__ gw,     // [E, H]
    float* __restrict__ combine)      // [T, E]
{
  int t = blockIdx.x * blockDim.x + threadIdx.x;
  if (t >= T_TOKENS) return;
  const float* xr = x + (size_t)t * HIDDEN;

  float s[N_EXPERTS];
  #pragma unroll
  for (int e = 0; e < N_EXPERTS; ++e) s[e] = 0.f;
  for (int h = 0; h < HIDDEN; ++h) {
    float xv = xr[h];
    #pragma unroll
    for (int e = 0; e < N_EXPERTS; ++e) s[e] += xv * gw[e * HIDDEN + h];
  }
  // softmax over 8 experts
  float mx = s[0];
  #pragma unroll
  for (int e = 1; e < N_EXPERTS; ++e) mx = fmaxf(mx, s[e]);
  float p[N_EXPERTS]; float sum = 0.f;
  #pragma unroll
  for (int e = 0; e < N_EXPERTS; ++e) { p[e] = __expf(s[e] - mx); sum += p[e]; }
  float inv = 1.f / sum;
  #pragma unroll
  for (int e = 0; e < N_EXPERTS; ++e) p[e] *= inv;

  // group scores: 4 groups of 2, keep top-2 groups (ties -> lowest index)
  float gs[4];
  #pragma unroll
  for (int g = 0; g < 4; ++g) gs[g] = fmaxf(p[2 * g], p[2 * g + 1]);
  int g0 = 0;
  #pragma unroll
  for (int g = 1; g < 4; ++g) if (gs[g] > gs[g0]) g0 = g;
  int g1 = (g0 == 0) ? 1 : 0;
  #pragma unroll
  for (int g = 0; g < 4; ++g) if (g != g0 && gs[g] > gs[g1]) g1 = g;

  float ms[N_EXPERTS];
  #pragma unroll
  for (int e = 0; e < N_EXPERTS; ++e) {
    int g = e >> 1;
    ms[e] = (g == g0 || g == g1) ? p[e] : 0.f;
  }
  // top-2 experts among masked scores
  int e0 = 0;
  #pragma unroll
  for (int e = 1; e < N_EXPERTS; ++e) if (ms[e] > ms[e0]) e0 = e;
  int e1 = (e0 == 0) ? 1 : 0;
  #pragma unroll
  for (int e = 0; e < N_EXPERTS; ++e) if (e != e0 && ms[e] > ms[e1]) e1 = e;

  float winv = 1.f / (ms[e0] + ms[e1]);
  #pragma unroll
  for (int e = 0; e < N_EXPERTS; ++e) {
    float cv = 0.f;
    if (e == e0 || e == e1) cv = ms[e] * winv;
    combine[(size_t)t * N_EXPERTS + e] = cv;
  }
}

// ---------------------------------------------------------------------------
// Kernel 2: streaming fp32 -> bf16 conversion (8 elements / thread)
// ---------------------------------------------------------------------------
__global__ __launch_bounds__(256) void cvt_kernel(
    const float* __restrict__ src, bf16_t* __restrict__ dst, int n)
{
  int i = (blockIdx.x * 256 + threadIdx.x) * 8;
  if (i >= n) return;
  float4 f0 = *(const float4*)(src + i);
  float4 f1 = *(const float4*)(src + i + 4);
  union { bf16_t h[8]; uint4 u; } o;
  o.h[0] = (bf16_t)f0.x; o.h[1] = (bf16_t)f0.y;
  o.h[2] = (bf16_t)f0.z; o.h[3] = (bf16_t)f0.w;
  o.h[4] = (bf16_t)f1.x; o.h[5] = (bf16_t)f1.y;
  o.h[6] = (bf16_t)f1.z; o.h[7] = (bf16_t)f1.w;
  *(uint4*)(dst + i) = o.u;
}

// ---------------------------------------------------------------------------
// TDM: DMA a 16-row x 1024-col bf16 tile (row stride 1024 elems) into LDS.
// D# built per CDNA5 ISA ch.8 (group0: count/lds_addr/global_addr/type,
// group1: data_size=2B, tensor_dim0=1024, tensor_dim1=16, tile 1024x16,
// dim0_stride=1024). Groups 2/3 zero (2-D tensor).
// ---------------------------------------------------------------------------
__device__ __forceinline__ void tdm_load_x_tile(const bf16_t* gsrc,
                                                unsigned int lds_off) {
  unsigned long long ga = (unsigned long long)(uintptr_t)gsrc;
  v4u g0;
  g0.x = 1u;                                               // count=1 (user D#)
  g0.y = lds_off;                                          // lds_addr (bytes)
  g0.z = (unsigned int)(ga & 0xFFFFFFFFu);                 // global_addr[31:0]
  g0.w = (unsigned int)((ga >> 32) & 0x1FFFFFFu) | (2u << 30); // [56:32]|type=2
  v8i g1;
  g1[0] = (int)(1u << 16);          // workgroup_mask=0, data_size=1 (2 bytes)
  g1[1] = (int)(1024u << 16);       // tensor_dim0[15:0]=1024 (bits 48..63)
  g1[2] = (int)(16u << 16);         // tensor_dim0[31:16]=0, tensor_dim1 lo=16
  g1[3] = (int)(1024u << 16);       // tensor_dim1 hi=0, tile_dim0=1024
  g1[4] = 16;                       // tile_dim1=16, tile_dim2=0
  g1[5] = 1024;                     // tensor_dim0_stride[31:0]=1024
  g1[6] = 0;                        // stride hi, tensor_dim1_stride lo
  g1[7] = 0;
  v4i z4 = {0, 0, 0, 0};
#if __clang_major__ >= 23
  v8i z8 = {0, 0, 0, 0, 0, 0, 0, 0};
  __builtin_amdgcn_tensor_load_to_lds(g0, g1, z4, z4, z8, 0);
#else
  __builtin_amdgcn_tensor_load_to_lds(g0, g1, z4, z4, 0);
#endif
}

// ---------------------------------------------------------------------------
// Kernel 3: fused expert MLP.
// Block = 256 threads = 8 wave32. blockIdx.x = 16-token tile, blockIdx.y = expert.
// X tile staged into LDS by the Tensor Data Mover (wave 0), synced with
// s_wait_tensorcnt + workgroup barrier.
// Pass 1: gu = x @ w13[e]^T via v_wmma_f32_16x16x32_bf16, silu*up*combine
//         -> ACTs[16][2048] bf16 in LDS.
// Pass 2: out += ACTs @ w2[e]^T via WMMA, accumulated with global f32 atomics.
// ---------------------------------------------------------------------------
__global__ __launch_bounds__(256) void moe_expert_kernel(
    const bf16_t* __restrict__ Xb,     // [T, H] bf16
    const bf16_t* __restrict__ W13b,   // [E, 2I, H] bf16
    const bf16_t* __restrict__ W2b,    // [E, H, I] bf16
    const float*  __restrict__ combine,// [T, E]
    float* __restrict__ out)           // [T, H] f32 (pre-zeroed)
{
  __shared__ __align__(16) bf16_t Xs[16 * HIDDEN];    // 32 KB
  __shared__ __align__(16) bf16_t ACTs[16 * INTER];   // 64 KB
  __shared__ float Cw[16];

  const int m0    = blockIdx.x * 16;
  const int e     = blockIdx.y;
  const int wave  = threadIdx.x >> 5;
  const int lane  = threadIdx.x & 31;
  const int col   = lane & 15;   // A-row / B-col / C-col index for this lane
  const int khalf = lane >> 4;   // which K-half this lane holds

  // wave 0: kick TDM for the x tile, load combine weights, wait for DMA
  if (wave == 0) {
    tdm_load_x_tile(Xb + (size_t)m0 * HIDDEN,
                    (unsigned int)(uintptr_t)&Xs[0]);
    if (threadIdx.x < 16)
      Cw[threadIdx.x] = combine[(size_t)(m0 + threadIdx.x) * N_EXPERTS + e];
    __builtin_amdgcn_s_wait_tensorcnt(0);
  }
  __syncthreads();

  union Frag { v16bf v; uint4 u[2]; };

  // ---------------- Pass 1: gate/up GEMM + SwiGLU ----------------
  for (int p = wave; p < INTER / 16; p += 8) {       // uniform per wave
    const int nb = p * 16;
    const bf16_t* bg = W13b + ((size_t)e * (2 * INTER) + nb + col) * HIDDEN;
    const bf16_t* bu = bg + (size_t)INTER * HIDDEN;
    const bf16_t* ar = Xs + (size_t)col * HIDDEN;

    v8f cg = {0.f, 0.f, 0.f, 0.f, 0.f, 0.f, 0.f, 0.f};
    v8f cu = {0.f, 0.f, 0.f, 0.f, 0.f, 0.f, 0.f, 0.f};
    #pragma unroll 4
    for (int k0 = 0; k0 < HIDDEN; k0 += 32) {
      Frag a, b;
      a.u[0] = *(const uint4*)(ar + k0 + khalf * 8);
      a.u[1] = *(const uint4*)(ar + k0 + 16 + khalf * 8);
      b.u[0] = *(const uint4*)(bg + k0 + khalf * 8);
      b.u[1] = *(const uint4*)(bg + k0 + 16 + khalf * 8);
      cg = __builtin_amdgcn_wmma_f32_16x16x32_bf16(false, a.v, false, b.v,
                                                   (short)0, cg, false, false);
      b.u[0] = *(const uint4*)(bu + k0 + khalf * 8);
      b.u[1] = *(const uint4*)(bu + k0 + 16 + khalf * 8);
      cu = __builtin_amdgcn_wmma_f32_16x16x32_bf16(false, a.v, false, b.v,
                                                   (short)0, cu, false, false);
    }
    #pragma unroll
    for (int r = 0; r < 8; ++r) {
      int row = r + 8 * khalf;
      float g = cg[r], u = cu[r];
      float act = (g / (1.f + __expf(-g))) * u * Cw[row];
      ACTs[(size_t)row * INTER + nb + col] = (bf16_t)act;
    }
  }
  __syncthreads();

  // ---------------- Pass 2: down-projection GEMM ----------------
  for (int q = wave; q < HIDDEN / 16; q += 8) {      // uniform per wave
    const int hb = q * 16;
    const bf16_t* bw = W2b + ((size_t)e * HIDDEN + hb + col) * INTER;
    const bf16_t* ar = ACTs + (size_t)col * INTER;

    v8f c = {0.f, 0.f, 0.f, 0.f, 0.f, 0.f, 0.f, 0.f};
    #pragma unroll 4
    for (int k0 = 0; k0 < INTER; k0 += 32) {
      Frag a, b;
      a.u[0] = *(const uint4*)(ar + k0 + khalf * 8);
      a.u[1] = *(const uint4*)(ar + k0 + 16 + khalf * 8);
      b.u[0] = *(const uint4*)(bw + k0 + khalf * 8);
      b.u[1] = *(const uint4*)(bw + k0 + 16 + khalf * 8);
      c = __builtin_amdgcn_wmma_f32_16x16x32_bf16(false, a.v, false, b.v,
                                                  (short)0, c, false, false);
    }
    #pragma unroll
    for (int r = 0; r < 8; ++r) {
      int row = r + 8 * khalf;
      float* dst = out + (size_t)(m0 + row) * HIDDEN + hb + col;
      __hip_atomic_fetch_add(dst, c[r], __ATOMIC_RELAXED, __HIP_MEMORY_SCOPE_AGENT);
    }
  }
}

// ---------------------------------------------------------------------------
// Launch
// ---------------------------------------------------------------------------
extern "C" void kernel_launch(void* const* d_in, const int* in_sizes, int n_in,
                              void* d_out, int out_size, void* d_ws, size_t ws_size,
                              hipStream_t stream) {
  const float* x   = (const float*)d_in[0];  // [1,1024,1024]
  const float* gw  = (const float*)d_in[1];  // [8,1024]
  const float* w13 = (const float*)d_in[2];  // [8,4096,1024]
  const float* w2  = (const float*)d_in[3];  // [8,1024,2048]
  float* out = (float*)d_out;

  // workspace layout
  char* w = (char*)d_ws;
  float*  combine = (float*)w;                                   // 32 KB
  bf16_t* Xb   = (bf16_t*)(w + (32 << 10));                      // 2 MB
  bf16_t* W13b = (bf16_t*)(w + (32 << 10) + (2 << 20));          // 64 MB
  bf16_t* W2b  = W13b + (size_t)N_EXPERTS * 2 * INTER * HIDDEN;  // 32 MB

  const int nX   = T_TOKENS * HIDDEN;                 // 1,048,576
  const int nW13 = N_EXPERTS * 2 * INTER * HIDDEN;    // 33,554,432
  const int nW2  = N_EXPERTS * HIDDEN * INTER;        // 16,777,216

  hipMemsetAsync(d_out, 0, (size_t)out_size * sizeof(float), stream);

  gate_kernel<<<T_TOKENS / 256, 256, 0, stream>>>(x, gw, combine);
  cvt_kernel<<<nX   / (8 * 256), 256, 0, stream>>>(x,   Xb,   nX);
  cvt_kernel<<<nW13 / (8 * 256), 256, 0, stream>>>(w13, W13b, nW13);
  cvt_kernel<<<nW2  / (8 * 256), 256, 0, stream>>>(w2,  W2b,  nW2);

  dim3 grid(T_TOKENS / 16, N_EXPERTS);
  moe_expert_kernel<<<grid, 256, 0, stream>>>(Xb, W13b, W2b, combine, out);
}
